// GLUMLP_41068477285033
// MI455X (gfx1250) — matrix operand order repaired and checked by reading, not compiled, not run on hardware
//
#include <hip/hip_runtime.h>
#include <hip/hip_bf16.h>

#define NUM_E   8
#define DIM_D   1024
#define DIM_H   1024
#define DIM_2H  2048
#define TOPK    2
#define MT      64      // token-slots per row tile (4 WMMA M sub-blocks)
#define CG      256     // columns per block (column-group)
#define RS      2080    // LDS row stride bytes: 2048 + 2x16B TDM pad (pad every 1024B)

typedef __attribute__((ext_vector_type(16))) __bf16 v16bf;
typedef __attribute__((ext_vector_type(8)))  __bf16 v8bf;
typedef __attribute__((ext_vector_type(8)))  float  v8f;
typedef __attribute__((ext_vector_type(4)))  unsigned int v4u;
typedef __attribute__((ext_vector_type(8)))  int v8i;
typedef __attribute__((ext_vector_type(4)))  int v4i;

static __device__ __forceinline__ v16bf cat16(v8bf lo, v8bf hi) {
  return __builtin_shufflevector(lo, hi, 0,1,2,3,4,5,6,7,8,9,10,11,12,13,14,15);
}

// weight element load (f32 path converts, bf16 path is a raw 16-bit load)
static __device__ __forceinline__ __bf16 ldw(const float* p)  { return (__bf16)*p; }
static __device__ __forceinline__ __bf16 ldw(const __bf16* p) { return *p; }

// LDS address of element c (bf16) in row `row` under the TDM-padded layout:
// 16B pad inserted after every 1024B stored -> +16B for c >= 512, row stride 2080B.
static __device__ __forceinline__ const v8bf* lds_chunk(const char* b, int row, int c) {
  return (const v8bf*)(b + row * RS + 2 * c + ((c >> 9) << 4));
}

// ---- TDM gather: 16 rows of `rowElems` bf16, rows picked by 16-bit indices ----
// D# packing per CDNA5 ISA 8.3-8.7 (gather mode, data_size=2B, LDS padding on).
static __device__ __forceinline__ void tdm_gather16(const void* gbase, unsigned ldsByte,
                                                    int rowElems, int nRows,
                                                    const unsigned* ip /* 8 packed idx pairs */) {
  const unsigned long long ga = (unsigned long long)gbase;
  v4u g0;
  g0[0] = 0x80000001u;                                   // count=1 | gather_mode | 16b idx
  g0[1] = ldsByte;                                       // lds_addr
  g0[2] = (unsigned)ga;                                  // global_addr[31:0]
  g0[3] = (unsigned)((ga >> 32) & 0x01ffffffu) | 0x80000000u; // global_addr[56:32] | type=2
  v8i g1;
  g1[0] = (1 << 16)                                      // data_size = 2 bytes
        | (1 << 20)                                      // pad_enable
        | (7 << 22)                                      // pad_interval = 256 dwords (1024B)
        | (3 << 25);                                     // pad_amount = 4 dwords (16B)
  g1[1] = (rowElems & 0xffff) << 16;                     // abar=0 | tensor_dim0 lo
  g1[2] = ((rowElems >> 16) & 0xffff) | ((nRows & 0xffff) << 16);  // td0 hi | td1 lo
  g1[3] = ((nRows >> 16) & 0xffff) | ((rowElems & 0xffff) << 16);  // td1 hi | tile_dim0
  g1[4] = 16;                                            // tile_dim1 = 16 indices
  g1[5] = rowElems;                                      // tensor_dim0_stride lo
  g1[6] = 0;
  g1[7] = 0;
  v4i g2, g3;
  g2[0] = (int)ip[0]; g2[1] = (int)ip[1]; g2[2] = (int)ip[2]; g2[3] = (int)ip[3];
  g3[0] = (int)ip[4]; g3[1] = (int)ip[5]; g3[2] = (int)ip[6]; g3[3] = (int)ip[7];
  const v8i g4 = {0, 0, 0, 0, 0, 0, 0, 0};               // unused extra descriptor group
  __builtin_amdgcn_tensor_load_to_lds(g0, g1, g2, g3, g4, 0);
}

// ---------------------------------------------------------------- router
__global__ void router(const int* __restrict__ idxs, int nk,
                       int* __restrict__ counts, int* __restrict__ lists) {
  int t = threadIdx.x;
  if (t < NUM_E) counts[t] = 0;
  __syncthreads();
  for (int s = t; s < nk; s += blockDim.x) {
    int e = idxs[s];
    int pos = atomicAdd(&counts[e], 1);
    lists[e * nk + pos] = s;
  }
}

// ---------------------------------------------------------------- f32 -> bf16 copy
__global__ void cvt_bf16(const float* __restrict__ in, __bf16* __restrict__ out, int n4) {
  int i = blockIdx.x * blockDim.x + threadIdx.x;
  if (i >= n4) return;
  float4 v = ((const float4*)in)[i];
  __bf16* d = out + (size_t)i * 4;
  d[0] = (__bf16)v.x; d[1] = (__bf16)v.y; d[2] = (__bf16)v.z; d[3] = (__bf16)v.w;
}

// ---- shared staging: 4 TDM gathers (waves 0..3), then barrier ----
static __device__ __forceinline__ void stage_tile_tdm(const void* gbase, int nRows,
                                                      const int* list, int tileBase, int count,
                                                      int wave, int shiftTok, char* smem) {
  if (wave < 4) {
    const int rbase = tileBase + wave * 16;
    unsigned ip[8];
    #pragma unroll
    for (int j = 0; j < 8; ++j) {
      const int sl0 = rbase + 2 * j, sl1 = sl0 + 1;
      const int s0 = (sl0 < count) ? list[sl0] : list[tileBase];
      const int s1 = (sl1 < count) ? list[sl1] : list[tileBase];
      const unsigned r0 = (unsigned)(s0 >> shiftTok);
      const unsigned r1 = (unsigned)(s1 >> shiftTok);
      ip[j] = (r0 & 0xffffu) | (r1 << 16);
    }
    tdm_gather16(gbase, (unsigned)(uintptr_t)smem + (unsigned)(wave * 16 * RS),
                 DIM_H, nRows, ip);
    __builtin_amdgcn_s_wait_tensorcnt(0);
  }
  __syncthreads();
}

// ---------------------------------------------------------------- GEMM1 + SiLU gate
template <typename WT>
__global__ __launch_bounds__(256)
void moe_gemm1_t(const __bf16* __restrict__ xb,
                 const WT*     __restrict__ W1,
                 const int*    __restrict__ counts,
                 const int*    __restrict__ lists,
                 __bf16*       __restrict__ a_act,
                 int nk, int nTok)
{
  const int e        = blockIdx.y;
  const int count    = counts[e];
  const int tileBase = blockIdx.x * MT;
  if (tileBase >= count) return;
  const int colBase  = blockIdx.z * CG;
  const int* list = lists + (size_t)e * nk;

  extern __shared__ char smem[];
  const int tid  = threadIdx.x;
  const int lane = tid & 31;
  const int wave = tid >> 5;
  const int ln   = lane & 15;
  const int half = lane >> 4;

  stage_tile_tdm(xb, nTok, list, tileBase, count, wave, /*shiftTok=*/1, smem);

  const WT* W1e = W1 + (size_t)e * DIM_D * DIM_2H;
  const bool fullTile = (tileBase + MT <= count);
  const v8f vzero = {};

  for (int s = wave; s < CG / 16; s += 8) {
    const int col0 = colBase + s * 16;
    v8f ah[4], ag[4];
    #pragma unroll
    for (int rb = 0; rb < 4; ++rb) { ah[rb] = vzero; ag[rb] = vzero; }

    const WT* bBase = W1e + col0 + ln + (size_t)(8 * half) * DIM_2H;
    for (int kk = 0; kk < DIM_D; kk += 32) {
      const WT* pB = bBase + (size_t)kk * DIM_2H;
      __builtin_prefetch(pB + (size_t)32 * DIM_2H, 0, 1);
      v16bf bh, bg;
      #pragma unroll
      for (int i = 0; i < 16; ++i) {
        bh[i] = ldw(pB + (size_t)(i < 8 ? i : i + 8) * DIM_2H);
        bg[i] = ldw(pB + (size_t)(i < 8 ? i : i + 8) * DIM_2H + DIM_H);
      }
      const int kLo = kk + 8 * half;
      #pragma unroll
      for (int rb = 0; rb < 4; ++rb) {
        v8bf lo = *lds_chunk(smem, rb * 16 + ln, kLo);
        v8bf hi = *lds_chunk(smem, rb * 16 + ln, kLo + 16);
        v16bf a = cat16(lo, hi);
        ah[rb] = __builtin_amdgcn_wmma_f32_16x16x32_bf16(false, a, false, bh, (short)0, ah[rb], false, false);
        ag[rb] = __builtin_amdgcn_wmma_f32_16x16x32_bf16(false, a, false, bg, (short)0, ag[rb], false, false);
      }
    }
    #pragma unroll
    for (int rb = 0; rb < 4; ++rb) {
      #pragma unroll
      for (int r = 0; r < 8; ++r) {
        const int sl = tileBase + rb * 16 + 8 * half + r;
        if (fullTile || sl < count) {
          const int slot = list[sl];
          const float g = ag[rb][r];
          const float a = ah[rb][r] * g * __builtin_amdgcn_rcpf(1.0f + __expf(-g));
          a_act[(size_t)slot * DIM_H + col0 + ln] = (__bf16)a;
        }
      }
    }
  }
}

// ---------------------------------------------------------------- GEMM2
template <typename WT>
__global__ __launch_bounds__(256)
void moe_gemm2_t(const __bf16* __restrict__ a_act,
                 const float*  __restrict__ gate_p,
                 const WT*     __restrict__ W2,
                 const int*    __restrict__ counts,
                 const int*    __restrict__ lists,
                 float*        __restrict__ y_pair,
                 int nk)
{
  const int e        = blockIdx.y;
  const int count    = counts[e];
  const int tileBase = blockIdx.x * MT;
  if (tileBase >= count) return;
  const int colBase  = blockIdx.z * CG;
  const int* list = lists + (size_t)e * nk;

  extern __shared__ char smem[];
  const int tid  = threadIdx.x;
  const int lane = tid & 31;
  const int wave = tid >> 5;
  const int ln   = lane & 15;
  const int half = lane >> 4;

  stage_tile_tdm(a_act, nk, list, tileBase, count, wave, /*shiftTok=*/0, smem);

  const WT* W2e = W2 + (size_t)e * DIM_H * DIM_D;
  const bool fullTile = (tileBase + MT <= count);
  const v8f vzero = {};

  for (int s = wave; s < CG / 16; s += 8) {
    const int col0 = colBase + s * 16;
    v8f c[4];
    #pragma unroll
    for (int rb = 0; rb < 4; ++rb) c[rb] = vzero;

    const WT* bBase = W2e + col0 + ln + (size_t)(8 * half) * DIM_D;
    for (int kk = 0; kk < DIM_H; kk += 32) {
      const WT* pB = bBase + (size_t)kk * DIM_D;
      __builtin_prefetch(pB + (size_t)32 * DIM_D, 0, 1);
      v16bf b;
      #pragma unroll
      for (int i = 0; i < 16; ++i) b[i] = ldw(pB + (size_t)(i < 8 ? i : i + 8) * DIM_D);
      const int kLo = kk + 8 * half;
      #pragma unroll
      for (int rb = 0; rb < 4; ++rb) {
        v8bf lo = *lds_chunk(smem, rb * 16 + ln, kLo);
        v8bf hi = *lds_chunk(smem, rb * 16 + ln, kLo + 16);
        v16bf a = cat16(lo, hi);
        c[rb] = __builtin_amdgcn_wmma_f32_16x16x32_bf16(false, a, false, b, (short)0, c[rb], false, false);
      }
    }
    #pragma unroll
    for (int rb = 0; rb < 4; ++rb) {
      #pragma unroll
      for (int r = 0; r < 8; ++r) {
        const int sl = tileBase + rb * 16 + 8 * half + r;
        if (fullTile || sl < count) {
          const int slot = list[sl];
          y_pair[(size_t)slot * DIM_D + col0 + ln] = gate_p[slot] * c[rb][r];
        }
      }
    }
  }
}

// ---------------------------------------------------------------- combine top-k pairs
__global__ void combine(const float* __restrict__ y_pair,
                        float* __restrict__ out, int n4) {
  const int i = blockIdx.x * blockDim.x + threadIdx.x;
  if (i >= n4) return;
  const int t  = i >> 8;                 // DIM_D/4 == 256
  const int c4 = i & 255;
  const float4 a = ((const float4*)y_pair)[(size_t)(2 * t)     * 256 + c4];
  const float4 b = ((const float4*)y_pair)[(size_t)(2 * t + 1) * 256 + c4];
  ((float4*)out)[i] = make_float4(a.x + b.x, a.y + b.y, a.z + b.z, a.w + b.w);
}

// ---------------------------------------------------------------- launcher
extern "C" void kernel_launch(void* const* d_in, const int* in_sizes, int n_in,
                              void* d_out, int out_size, void* d_ws, size_t ws_size,
                              hipStream_t stream)
{
  const float* x   = (const float*)d_in[0];
  const float* p   = (const float*)d_in[1];
  const int*   idx = (const int*)d_in[2];
  const float* W1  = (const float*)d_in[3];
  const float* W2  = (const float*)d_in[4];
  float* out = (float*)d_out;

  const int N  = in_sizes[0] / DIM_D;   // tokens
  const int nk = N * TOPK;              // token-slots

  // workspace layout (256B-aligned sections)
  char* ws = (char*)d_ws;
  int*    counts = (int*)ws;                                    // 8 ints
  int*    lists  = (int*)(ws + 256);                            // E*nk ints
  size_t  off    = 256 + (size_t)NUM_E * nk * sizeof(int);
  off            = (off + 255) & ~(size_t)255;
  __bf16* xb     = (__bf16*)(ws + off);                         // N*D bf16 (4MB)
  off           += (size_t)N * DIM_D * sizeof(__bf16);
  off            = (off + 255) & ~(size_t)255;
  __bf16* a_act  = (__bf16*)(ws + off);                         // nk*H bf16 (8MB)
  off           += (size_t)nk * DIM_H * sizeof(__bf16);
  off            = (off + 255) & ~(size_t)255;
  float*  y_pair = (float*)(ws + off);                          // nk*D f32 (16MB)
  off           += (size_t)nk * DIM_D * sizeof(float);
  off            = (off + 255) & ~(size_t)255;

  // optional bf16 weight copies (48MB) if workspace allows
  const size_t w1e = (size_t)NUM_E * DIM_D * DIM_2H;            // W1 elements
  const size_t w2e = (size_t)NUM_E * DIM_H * DIM_D;             // W2 elements
  const bool bf16W = (ws_size >= off + (w1e + w2e) * sizeof(__bf16));
  __bf16* W1b = (__bf16*)(ws + off);
  __bf16* W2b = W1b + w1e;

  router<<<dim3(1), dim3(256), 0, stream>>>(idx, nk, counts, lists);

  const int nx4 = N * DIM_D / 4;
  cvt_bf16<<<dim3((nx4 + 255) / 256), dim3(256), 0, stream>>>(x, xb, nx4);

  const size_t smem = (size_t)MT * RS;                          // 133,120 B
  (void)hipFuncSetAttribute(reinterpret_cast<const void*>(moe_gemm1_t<float>),
                            hipFuncAttributeMaxDynamicSharedMemorySize, (int)smem);
  (void)hipFuncSetAttribute(reinterpret_cast<const void*>(moe_gemm2_t<float>),
                            hipFuncAttributeMaxDynamicSharedMemorySize, (int)smem);
  (void)hipFuncSetAttribute(reinterpret_cast<const void*>(moe_gemm1_t<__bf16>),
                            hipFuncAttributeMaxDynamicSharedMemorySize, (int)smem);
  (void)hipFuncSetAttribute(reinterpret_cast<const void*>(moe_gemm2_t<__bf16>),
                            hipFuncAttributeMaxDynamicSharedMemorySize, (int)smem);

  const int maxTiles = (nk + MT - 1) / MT;
  dim3 grid1(maxTiles, NUM_E, DIM_H / CG);
  dim3 grid2(maxTiles, NUM_E, DIM_D / CG);

  if (bf16W) {
    const int nw1 = (int)(w1e / 4), nw2 = (int)(w2e / 4);
    cvt_bf16<<<dim3((nw1 + 255) / 256), dim3(256), 0, stream>>>(W1, W1b, nw1);
    cvt_bf16<<<dim3((nw2 + 255) / 256), dim3(256), 0, stream>>>(W2, W2b, nw2);
    moe_gemm1_t<__bf16><<<grid1, dim3(256), smem, stream>>>(xb, W1b, counts, lists, a_act, nk, N);
    moe_gemm2_t<__bf16><<<grid2, dim3(256), smem, stream>>>(a_act, p, W2b, counts, lists, y_pair, nk);
  } else {
    moe_gemm1_t<float><<<grid1, dim3(256), smem, stream>>>(xb, W1, counts, lists, a_act, nk, N);
    moe_gemm2_t<float><<<grid2, dim3(256), smem, stream>>>(a_act, p, W2, counts, lists, y_pair, nk);
  }

  const int n4 = out_size / 4;
  combine<<<dim3((n4 + 255) / 256), dim3(256), 0, stream>>>(y_pair, out, n4);
  (void)n_in;
}
